// gnn_embedder2_47854525612047
// MI455X (gfx1250) — matrix operand order repaired and checked by reading, not compiled, not run on hardware
//
#include <hip/hip_runtime.h>

// ---------------------------------------------------------------------------
// GNN embedder for MI455X (gfx1250, wave32, WMMA f16 16x16x32, f32 accum)
//
//  k_edge (fused)  : EdgeMLP L1 -> L2 -> outer-product q-GEMM -> scatter,
//                    with Wq staged to LDS via async global->LDS copies
//                    (double-buffered, ASYNCcnt) shared by all waves in WG.
//  k_root          : x@Wroot + agg + bc1         (WMMA)
//  k_gemm_f16<K,N> : GAT projections             (WMMA)
//  attention/BN/pool/head: VALU + atomics (bandwidth-trivial stages)
// ---------------------------------------------------------------------------

#define N_NODES 10000
#define N_EDGES 160000
#define N_GRAPH 100

typedef _Float16 v16h __attribute__((ext_vector_type(16)));
typedef float    v8f  __attribute__((ext_vector_type(8)));

#define DEV static __device__ __forceinline__

DEV v8f wmma16(v16h a, v16h b, v8f c) {
  // D = A(16x32 f16) x B(32x16 f16) + C(16x16 f32)
  return __builtin_amdgcn_wmma_f32_16x16x32_f16(false, a, false, b,
                                                (short)0, c, false, false);
}

// A-fragment (16-bit, 16x32): lane half selects K-group; element i -> local K = 16*half + i
DEV int a_k(int i, int half) { return i + 16 * half; }

DEV v16h afrag_f32row(const float* row, int kbase, int klimit, int half) {
  union { v16h v; _Float16 h[16]; } u;
#pragma unroll
  for (int i = 0; i < 16; ++i) {
    int K = kbase + a_k(i, half);
    u.h[i] = (K < klimit) ? (_Float16)row[K] : (_Float16)0.0f;
  }
  return u.v;
}

DEV v16h afrag_f16row(const _Float16* row, int kbase, int klimit, int half) {
  union { v16h v; _Float16 h[16]; } u;
#pragma unroll
  for (int i = 0; i < 16; ++i) {
    int K = kbase + a_k(i, half);
    u.h[i] = (K < klimit) ? row[K] : (_Float16)0.0f;
  }
  return u.v;
}

// B fragments stored frag-linear: frag[fragIdx][lane][16 halfs] (32B aligned)
DEV v16h bfrag(const _Float16* fr, int fragIdx, int lane) {
  return *(const v16h*)(fr + (size_t)fragIdx * 512 + (size_t)lane * 16);
}

// CDNA5 async global->LDS copy (ASYNCcnt-tracked), GV addressing mode.
DEV void async_copy_b128(unsigned lds_off, const void* gptr) {
  unsigned long long ga = (unsigned long long)(uintptr_t)gptr;
  asm volatile("global_load_async_to_lds_b128 %0, %1, off"
               :: "v"(lds_off), "v"(ga) : "memory");
}
DEV void async_wait0() { asm volatile("s_wait_asynccnt 0x0" ::: "memory"); }

// order-preserving float->uint mapping for atomicMax on floats
DEV unsigned fmapu(float f) {
  unsigned u = __float_as_uint(f);
  return (u & 0x80000000u) ? ~u : (u | 0x80000000u);
}
DEV float funmap(unsigned u) {
  unsigned v = (u & 0x80000000u) ? (u & 0x7FFFFFFFu) : ~u;
  return __uint_as_float(v);
}

DEV float lrelu(float v) { return v > 0.0f ? v : 0.2f * v; }

// ---------------------------------------------------------------------------
// Weight fragment builders
// B-frag element mapping: N = ntile*16 + (lane&15), K = kstep*32 + i + 16*(lane>>4)
// ---------------------------------------------------------------------------
__global__ void k_build_frags(const float* __restrict__ W, _Float16* __restrict__ out,
                              int Ktot, int Ntot, int Ksteps, int Ntiles) {
  int idx = blockIdx.x * blockDim.x + threadIdx.x;
  int total = Ksteps * Ntiles * 512;
  if (idx >= total) return;
  int i = idx & 15, lane = (idx >> 4) & 31, frag = idx >> 9;
  int nt = frag % Ntiles, ks = frag / Ntiles;
  int K = ks * 32 + i + 16 * (lane >> 4);
  int Ncol = nt * 16 + (lane & 15);
  float v = (K < Ktot && Ncol < Ntot) ? W[(size_t)K * Ntot + Ncol] : 0.0f;
  out[idx] = (_Float16)v;
}

// Wq[(k*32+f), o] = We3[k*1024 + f*32 + o]; extra kstep 64 carries be3 (A scale 1)
__global__ void k_build_frags_wq(const float* __restrict__ We3,
                                 const float* __restrict__ be3,
                                 _Float16* __restrict__ out) {
  int idx = blockIdx.x * blockDim.x + threadIdx.x;
  int total = 65 * 2 * 512;
  if (idx >= total) return;
  int i = idx & 15, lane = (idx >> 4) & 31, frag = idx >> 9;
  int nt = frag & 1, ks = frag >> 1;
  int Kl = i + 16 * (lane >> 4);     // 0..31 within kstep
  int Ncol = nt * 16 + (lane & 15);  // 0..31
  float v;
  if (ks < 64) v = We3[(size_t)ks * 1024 + (size_t)Kl * 32 + Ncol];
  else         v = be3[(size_t)Kl * 32 + Ncol];
  out[idx] = (_Float16)v;
}

// ---------------------------------------------------------------------------
// Fused edge pipeline. One wave = 16 edges, block = 4 waves, grid = 2500 exact.
// Wq (65 ksteps x 2 ntiles x 512 halfs = 130KB) is staged to LDS in 5 chunks
// of <=16 ksteps (32KB), double-buffered, via async global->LDS B128 copies.
// ---------------------------------------------------------------------------
#define WQ_CHUNK 16                 // ksteps per staging chunk
#define WQ_CHUNK_HALFS (WQ_CHUNK * 2 * 512)

__global__ void __launch_bounds__(128)
k_edge(const float* __restrict__ x, const int* __restrict__ src,
       const int* __restrict__ dst, const float* __restrict__ ea,
       const float* __restrict__ be1, const float* __restrict__ be2,
       const _Float16* __restrict__ We1f, const _Float16* __restrict__ We2f,
       const _Float16* __restrict__ Wqf, float* __restrict__ agg) {
  __shared__ __align__(32) _Float16 t1h[4][16][128];
  __shared__ __align__(32) float    t2f[4][16][64];
  __shared__ __align__(32) _Float16 wqs[2][WQ_CHUNK_HALFS];   // 2 x 32KB

  int tid = threadIdx.x;
  int w = tid >> 5;
  int lane = tid & 31;
  int half = lane >> 4;
  int m = lane & 15;
  int tile = blockIdx.x * 4 + w;
  int e = tile * 16 + m;

  // --- kick off async staging of Wq chunk 0 (overlaps EdgeMLP L1/L2 WMMAs)
  {
    int n128 = WQ_CHUNK * 128;                       // b128 transfers in chunk
    unsigned base = (unsigned)(uintptr_t)&wqs[0][0];
    for (int t = tid; t < n128; t += 128)
      async_copy_b128(base + t * 16, Wqf + (size_t)t * 8);
  }

  // --- Layer 1: [16x16] @ [16,128], K padded to 32 with zeros
  v16h aEA = afrag_f32row(ea + (size_t)e * 16, 0, 16, half);
#pragma unroll
  for (int nt = 0; nt < 8; ++nt) {
    v8f c = {};
    c = wmma16(aEA, bfrag(We1f, nt, lane), c);
    int col = nt * 16 + (lane & 15);
    float b = be1[col];
#pragma unroll
    for (int v = 0; v < 8; ++v) {
      float r = c[v] + b;
      r = r > 0.0f ? r : 0.0f;
      t1h[w][v + 8 * half][col] = (_Float16)r;
    }
  }
  __syncthreads();

  // --- Layer 2: [16x128] @ [128,64]
#pragma unroll
  for (int nt = 0; nt < 4; ++nt) {
    v8f c = {};
#pragma unroll
    for (int ks = 0; ks < 4; ++ks) {
      v16h a = afrag_f16row(&t1h[w][m][0], ks * 32, 128, half);
      c = wmma16(a, bfrag(We2f, ks * 4 + nt, lane), c);
    }
    int col = nt * 16 + (lane & 15);
    float b = be2[col];
#pragma unroll
    for (int v = 0; v < 8; ++v) {
      float r = c[v] + b;
      r = r > 0.0f ? r : 0.0f;
      t2f[w][v + 8 * half][col] = r;
    }
  }
  __syncthreads();

  // --- q-GEMM: msg[16,32] = q[16,2048] @ Wq[2048,32]; q is never materialized.
  // kstep k of q equals t2[e,k] * y[e,0..31]; kstep 64 is the be3 term.
  int s = src[e];
  float yf[16];
#pragma unroll
  for (int i = 0; i < 16; ++i) yf[i] = x[(size_t)s * 32 + a_k(i, half)];

  async_wait0();       // chunk 0 resident
  __syncthreads();

  v8f acc0 = {}, acc1 = {};
  for (int c = 0; c < 5; ++c) {           // chunks: 16,16,16,16,1 ksteps
    // prefetch next chunk into the other buffer
    if (c < 4) {
      int nks = (c + 1 < 4) ? WQ_CHUNK : 1;
      int n128 = nks * 128;
      unsigned base = (unsigned)(uintptr_t)&wqs[(c + 1) & 1][0];
      const _Float16* gsrc = Wqf + (size_t)(c + 1) * WQ_CHUNK_HALFS;
      for (int t = tid; t < n128; t += 128)
        async_copy_b128(base + t * 16, gsrc + (size_t)t * 8);
    }
    // consume current chunk from LDS
    const _Float16* wq = &wqs[c & 1][0];
    int nks = (c < 4) ? WQ_CHUNK : 1;
    for (int k = 0; k < nks; ++k) {
      int kk = c * WQ_CHUNK + k;
      float tk = (kk < 64) ? t2f[w][m][kk] : 1.0f;
      union { v16h v; _Float16 h[16]; } u;
#pragma unroll
      for (int i = 0; i < 16; ++i) u.h[i] = (_Float16)(tk * yf[i]);
      acc0 = wmma16(u.v, bfrag(wq, k * 2 + 0, lane), acc0);
      acc1 = wmma16(u.v, bfrag(wq, k * 2 + 1, lane), acc1);
    }
    async_wait0();     // own async copies complete ...
    __syncthreads();   // ... and every wave's copies visible to the WG
  }

  // --- scatter msg rows into agg[dst]
#pragma unroll
  for (int v = 0; v < 8; ++v) {
    int mc = v + 8 * half;
    int d = dst[tile * 16 + mc];
    atomicAdd(&agg[(size_t)d * 32 + (lane & 15)], acc0[v]);
    atomicAdd(&agg[(size_t)d * 32 + 16 + (lane & 15)], acc1[v]);
  }
}

// ---------------------------------------------------------------------------
// z = x @ Wroot + agg + bc1   (one wave = 16 node rows)
// ---------------------------------------------------------------------------
__global__ void __launch_bounds__(256)
k_root(const float* __restrict__ x, const float* __restrict__ agg,
       const float* __restrict__ bc1, const _Float16* __restrict__ Wrootf,
       float* __restrict__ z, int ntiles) {
  int w = threadIdx.x >> 5, lane = threadIdx.x & 31;
  int half = lane >> 4, m = lane & 15;
  int tile = blockIdx.x * 8 + w;
  if (tile >= ntiles) return;
  int row = tile * 16 + m;
  v16h a = afrag_f32row(x + (size_t)row * 32, 0, 32, half);
#pragma unroll
  for (int nt = 0; nt < 2; ++nt) {
    v8f c = {};
    c = wmma16(a, bfrag(Wrootf, nt, lane), c);
    int col = nt * 16 + (lane & 15);
#pragma unroll
    for (int v = 0; v < 8; ++v) {
      int rc = tile * 16 + v + 8 * half;
      z[(size_t)rc * 32 + col] = c[v] + agg[(size_t)rc * 32 + col] + bc1[col];
    }
  }
}

// ---------------------------------------------------------------------------
// Generic WMMA GEMM (compile-time shape): out[N][Nout] = H(f16)[N][Kin] @ Wfrag
// A-fragment loaded once per k-step, reused across all N-tiles.
// ---------------------------------------------------------------------------
template <int Ksteps, int Ntiles>
__global__ void __launch_bounds__(256)
k_gemm_f16(const _Float16* __restrict__ H, int Kin,
           const _Float16* __restrict__ Wf, float* __restrict__ out,
           int Nout, int ntilesRows) {
  int w = threadIdx.x >> 5, lane = threadIdx.x & 31;
  int half = lane >> 4, m = lane & 15;
  int tile = blockIdx.x * 8 + w;
  if (tile >= ntilesRows) return;
  const _Float16* rp = H + (size_t)(tile * 16 + m) * Kin;
  v8f acc[Ntiles] = {};
#pragma unroll
  for (int ks = 0; ks < Ksteps; ++ks) {
    v16h a = afrag_f16row(rp, ks * 32, Kin, half);
#pragma unroll
    for (int nt = 0; nt < Ntiles; ++nt)
      acc[nt] = wmma16(a, bfrag(Wf, ks * Ntiles + nt, lane), acc[nt]);
  }
#pragma unroll
  for (int nt = 0; nt < Ntiles; ++nt) {
    int col = nt * 16 + (lane & 15);
#pragma unroll
    for (int v = 0; v < 8; ++v) {
      int rc = tile * 16 + v + 8 * half;
      out[(size_t)rc * Nout + col] = acc[nt][v];
    }
  }
}

// ---------------------------------------------------------------------------
// BatchNorm (training mode) helpers
// ---------------------------------------------------------------------------
__global__ void k_colstats(const float* __restrict__ M, int rows, int cols,
                           float* __restrict__ stats) {
  int col = blockIdx.x;
  float s = 0.0f, s2 = 0.0f;
  for (int r = threadIdx.x; r < rows; r += blockDim.x) {
    float v = M[(size_t)r * cols + col];
    s += v; s2 += v * v;
  }
  __shared__ float sh[256], sh2[256];
  sh[threadIdx.x] = s; sh2[threadIdx.x] = s2;
  __syncthreads();
  for (int st = 128; st > 0; st >>= 1) {
    if ((int)threadIdx.x < st) {
      sh[threadIdx.x] += sh[threadIdx.x + st];
      sh2[threadIdx.x] += sh2[threadIdx.x + st];
    }
    __syncthreads();
  }
  if (threadIdx.x == 0) { stats[col] = sh[0]; stats[cols + col] = sh2[0]; }
}

__global__ void k_bnrelu(const float* __restrict__ in, float* __restrict__ outf,
                         _Float16* __restrict__ outh, const float* __restrict__ stats,
                         int rows, int cols, int doRelu) {
  int idx = blockIdx.x * blockDim.x + threadIdx.x;
  if (idx >= rows * cols) return;
  int col = idx % cols;
  float mean = stats[col] / (float)rows;
  float var = stats[cols + col] / (float)rows - mean * mean;
  float r = (in[idx] - mean) * rsqrtf(var + 1e-5f);
  if (doRelu) r = r > 0.0f ? r : 0.0f;
  if (outf) outf[idx] = r;
  if (outh) outh[idx] = (_Float16)r;
}

// ---------------------------------------------------------------------------
// GAT attention pieces (edges + PyG-style self loops)
// ---------------------------------------------------------------------------
__global__ void k_scdots(const float* __restrict__ hW, const float* __restrict__ as_,
                         const float* __restrict__ ad_, float* __restrict__ scs,
                         float* __restrict__ scd, int F, int n) {
  int i = blockIdx.x * blockDim.x + threadIdx.x;
  if (i >= n) return;
  float a = 0.0f, b = 0.0f;
  for (int f = 0; f < F; ++f) {
    float v = hW[(size_t)i * F + f];
    a += v * as_[f]; b += v * ad_[f];
  }
  scs[i] = a; scd[i] = b;
}

__global__ void k_att_max(const float* __restrict__ scs, const float* __restrict__ scd,
                          const int* __restrict__ src, const int* __restrict__ dst,
                          unsigned* __restrict__ emax, int e, int n) {
  int t = blockIdx.x * blockDim.x + threadIdx.x;
  if (t >= e + n) return;
  int s, d;
  if (t < e) { s = src[t]; d = dst[t]; } else { s = t - e; d = t - e; }
  atomicMax(&emax[d], fmapu(lrelu(scs[s] + scd[d])));
}

__global__ void k_att_exp(const float* __restrict__ scs, const float* __restrict__ scd,
                          const int* __restrict__ src, const int* __restrict__ dst,
                          const unsigned* __restrict__ emax, float* __restrict__ ex,
                          float* __restrict__ exloop, float* __restrict__ esum,
                          int e, int n) {
  int t = blockIdx.x * blockDim.x + threadIdx.x;
  if (t >= e + n) return;
  int s, d;
  if (t < e) { s = src[t]; d = dst[t]; } else { s = t - e; d = t - e; }
  float v = lrelu(scs[s] + scd[d]);
  float xv = __expf(v - funmap(emax[d]));
  if (t < e) ex[t] = xv; else exloop[d] = xv;
  atomicAdd(&esum[d], xv);
}

__global__ void k_att_scatter(const float* __restrict__ hW, const int* __restrict__ src,
                              const int* __restrict__ dst, const float* __restrict__ ex,
                              const float* __restrict__ esum, float* __restrict__ out,
                              int F, int e) {
  int t = blockIdx.x * blockDim.x + threadIdx.x;
  if (t >= e) return;
  int s = src[t], d = dst[t];
  float wgt = ex[t] / (esum[d] + 1e-16f);
  for (int f = 0; f < F; ++f)
    atomicAdd(&out[(size_t)d * F + f], wgt * hW[(size_t)s * F + f]);
}

__global__ void k_att_self_bias(const float* __restrict__ hW, const float* __restrict__ exloop,
                                const float* __restrict__ esum, const float* __restrict__ bg,
                                float* __restrict__ out, int F, int n) {
  int i = blockIdx.x * blockDim.x + threadIdx.x;
  if (i >= n) return;
  float wgt = exloop[i] / (esum[i] + 1e-16f);
  for (int f = 0; f < F; ++f)
    out[(size_t)i * F + f] += wgt * hW[(size_t)i * F + f] + bg[f];
}

// ---------------------------------------------------------------------------
// Pooling, head, utilities
// ---------------------------------------------------------------------------
__global__ void k_pool(const float* __restrict__ h, const int* __restrict__ batch,
                       float* __restrict__ pool, float* __restrict__ cnt, int F, int n) {
  int i = blockIdx.x * blockDim.x + threadIdx.x;
  if (i >= n) return;
  int b = batch[i];
  atomicAdd(&cnt[b], 1.0f);
  for (int f = 0; f < F; ++f)
    atomicAdd(&pool[(size_t)b * F + f], h[(size_t)i * F + f]);
}

__global__ void k_gmean(const float* __restrict__ pool, const float* __restrict__ cnt,
                        float* __restrict__ g, int F, int b) {
  int idx = blockIdx.x * blockDim.x + threadIdx.x;
  if (idx >= b * F) return;
  g[idx] = pool[idx] / fmaxf(cnt[idx / F], 1.0f);
}

__global__ void k_fc(const float* __restrict__ A, const float* __restrict__ W,
                     const float* __restrict__ bias, float* __restrict__ out,
                     int rows, int K, int Nout) {
  int idx = blockIdx.x * blockDim.x + threadIdx.x;
  if (idx >= rows * Nout) return;
  int r = idx / Nout, n = idx % Nout;
  float acc = bias ? bias[n] : 0.0f;
  for (int k = 0; k < K; ++k) acc += A[(size_t)r * K + k] * W[(size_t)k * Nout + n];
  out[idx] = acc;
}

__global__ void k_zero(float* __restrict__ p, size_t n) {
  size_t i = (size_t)blockIdx.x * blockDim.x + threadIdx.x;
  size_t stride = (size_t)gridDim.x * blockDim.x;
  for (; i < n; i += stride) p[i] = 0.0f;
}

// ---------------------------------------------------------------------------
extern "C" void kernel_launch(void* const* d_in, const int* in_sizes, int n_in,
                              void* d_out, int out_size, void* d_ws, size_t ws_size,
                              hipStream_t stream) {
  const float* x     = (const float*)d_in[0];
  const int*   ei    = (const int*)d_in[1];
  const int*   src   = ei;
  const int*   dst   = ei + N_EDGES;
  const float* ea    = (const float*)d_in[2];
  const int*   batch = (const int*)d_in[3];
  const float* We1 = (const float*)d_in[4],  *be1 = (const float*)d_in[5];
  const float* We2 = (const float*)d_in[6],  *be2 = (const float*)d_in[7];
  const float* We3 = (const float*)d_in[8],  *be3 = (const float*)d_in[9];
  const float* Wroot = (const float*)d_in[10], *bc1 = (const float*)d_in[11];
  const float* Wg1 = (const float*)d_in[12], *as1 = (const float*)d_in[13];
  const float* ad1 = (const float*)d_in[14], *bg1 = (const float*)d_in[15];
  const float* Wg2 = (const float*)d_in[16], *as2 = (const float*)d_in[17];
  const float* ad2 = (const float*)d_in[18], *bg2 = (const float*)d_in[19];
  const float* Wf1 = (const float*)d_in[20], *bf1 = (const float*)d_in[21];
  const float* Wf2 = (const float*)d_in[22], *bf2 = (const float*)d_in[23];
  const float* Wf3 = (const float*)d_in[24], *bf3 = (const float*)d_in[25];

  char* w = (char*)d_ws;
  size_t off = 0;
  auto B16 = [&](size_t halfs) -> _Float16* {
    _Float16* p = (_Float16*)(w + off);
    off += ((halfs * 2 + 255) / 256) * 256;
    return p;
  };
  auto F32 = [&](size_t n) -> float* {
    float* p = (float*)(w + off);
    off += ((n * 4 + 255) / 256) * 256;
    return p;
  };

  _Float16* We1f  = B16(1 * 8 * 512);
  _Float16* We2f  = B16(4 * 4 * 512);
  _Float16* Wqf   = B16(65 * 2 * 512);
  _Float16* Wrootf= B16(1 * 2 * 512);
  _Float16* Wg1f  = B16(1 * 4 * 512);
  _Float16* Wg2f  = B16(2 * 8 * 512);

  float* agg   = F32((size_t)N_NODES * 32);
  float* z     = F32((size_t)N_NODES * 32);
  _Float16* h0h = B16((size_t)N_NODES * 32);
  float* hW1   = F32((size_t)N_NODES * 64);
  float* g1out = F32((size_t)N_NODES * 64);
  _Float16* h1h = B16((size_t)N_NODES * 64);
  float* hW2   = F32((size_t)N_NODES * 128);
  float* g2out = F32((size_t)N_NODES * 128);
  float* h2    = F32((size_t)N_NODES * 128);
  float* scs   = F32(N_NODES);
  float* scd   = F32(N_NODES);
  unsigned* emax = (unsigned*)F32(N_NODES);
  float* esum  = F32(N_NODES);
  float* exb   = F32(N_EDGES);
  float* exloop= F32(N_NODES);
  float* stats = F32(512);
  float* pool  = F32((size_t)N_GRAPH * 128);
  float* cnt   = F32(N_GRAPH);
  float* g     = F32((size_t)N_GRAPH * 128);
  float* t3    = F32((size_t)N_GRAPH * 256);
  float* t4    = F32((size_t)N_GRAPH * 128);

  auto cdiv = [](int a, int b) { return (a + b - 1) / b; };

  // --- build f16 weight fragments
  k_build_frags<<<cdiv(1 * 8 * 512, 256), 256, 0, stream>>>(We1, We1f, 16, 128, 1, 8);
  k_build_frags<<<cdiv(4 * 4 * 512, 256), 256, 0, stream>>>(We2, We2f, 128, 64, 4, 4);
  k_build_frags_wq<<<cdiv(65 * 2 * 512, 256), 256, 0, stream>>>(We3, be3, Wqf);
  k_build_frags<<<cdiv(1 * 2 * 512, 256), 256, 0, stream>>>(Wroot, Wrootf, 32, 32, 1, 2);
  k_build_frags<<<cdiv(1 * 4 * 512, 256), 256, 0, stream>>>(Wg1, Wg1f, 32, 64, 1, 4);
  k_build_frags<<<cdiv(2 * 8 * 512, 256), 256, 0, stream>>>(Wg2, Wg2f, 64, 128, 2, 8);

  // --- NNConv (fused edge pipeline) ---
  k_zero<<<256, 256, 0, stream>>>(agg, (size_t)N_NODES * 32);
  k_edge<<<2500, 128, 0, stream>>>(x, src, dst, ea, be1, be2, We1f, We2f, Wqf, agg);
  k_root<<<cdiv(625, 8), 256, 0, stream>>>(x, agg, bc1, Wrootf, z, 625);
  k_colstats<<<32, 256, 0, stream>>>(z, N_NODES, 32, stats);
  k_bnrelu<<<cdiv(N_NODES * 32, 256), 256, 0, stream>>>(z, (float*)nullptr, h0h,
                                                        stats, N_NODES, 32, 1);

  // --- GAT layer 1 (32 -> 64) ---
  k_gemm_f16<1, 4><<<cdiv(625, 8), 256, 0, stream>>>(h0h, 32, Wg1f, hW1, 64, 625);
  k_scdots<<<cdiv(N_NODES, 256), 256, 0, stream>>>(hW1, as1, ad1, scs, scd, 64, N_NODES);
  k_zero<<<64, 256, 0, stream>>>((float*)emax, N_NODES);
  k_zero<<<64, 256, 0, stream>>>(esum, N_NODES);
  k_zero<<<256, 256, 0, stream>>>(g1out, (size_t)N_NODES * 64);
  k_att_max<<<cdiv(N_EDGES + N_NODES, 256), 256, 0, stream>>>(scs, scd, src, dst, emax,
                                                              N_EDGES, N_NODES);
  k_att_exp<<<cdiv(N_EDGES + N_NODES, 256), 256, 0, stream>>>(scs, scd, src, dst, emax,
                                                              exb, exloop, esum,
                                                              N_EDGES, N_NODES);
  k_att_scatter<<<cdiv(N_EDGES, 256), 256, 0, stream>>>(hW1, src, dst, exb, esum,
                                                        g1out, 64, N_EDGES);
  k_att_self_bias<<<cdiv(N_NODES, 256), 256, 0, stream>>>(hW1, exloop, esum, bg1,
                                                          g1out, 64, N_NODES);
  k_colstats<<<64, 256, 0, stream>>>(g1out, N_NODES, 64, stats);
  k_bnrelu<<<cdiv(N_NODES * 64, 256), 256, 0, stream>>>(g1out, (float*)nullptr, h1h,
                                                        stats, N_NODES, 64, 1);

  // --- GAT layer 2 (64 -> 128) ---
  k_gemm_f16<2, 8><<<cdiv(625, 8), 256, 0, stream>>>(h1h, 64, Wg2f, hW2, 128, 625);
  k_scdots<<<cdiv(N_NODES, 256), 256, 0, stream>>>(hW2, as2, ad2, scs, scd, 128, N_NODES);
  k_zero<<<64, 256, 0, stream>>>((float*)emax, N_NODES);
  k_zero<<<64, 256, 0, stream>>>(esum, N_NODES);
  k_zero<<<256, 256, 0, stream>>>(g2out, (size_t)N_NODES * 128);
  k_att_max<<<cdiv(N_EDGES + N_NODES, 256), 256, 0, stream>>>(scs, scd, src, dst, emax,
                                                              N_EDGES, N_NODES);
  k_att_exp<<<cdiv(N_EDGES + N_NODES, 256), 256, 0, stream>>>(scs, scd, src, dst, emax,
                                                              exb, exloop, esum,
                                                              N_EDGES, N_NODES);
  k_att_scatter<<<cdiv(N_EDGES, 256), 256, 0, stream>>>(hW2, src, dst, exb, esum,
                                                        g2out, 128, N_EDGES);
  k_att_self_bias<<<cdiv(N_NODES, 256), 256, 0, stream>>>(hW2, exloop, esum, bg2,
                                                          g2out, 128, N_NODES);
  k_colstats<<<128, 256, 0, stream>>>(g2out, N_NODES, 128, stats);
  k_bnrelu<<<cdiv(N_NODES * 128, 256), 256, 0, stream>>>(g2out, h2, (_Float16*)nullptr,
                                                         stats, N_NODES, 128, 1);

  // --- global mean pool ---
  k_zero<<<64, 256, 0, stream>>>(pool, (size_t)N_GRAPH * 128);
  k_zero<<<1, 128, 0, stream>>>(cnt, N_GRAPH);
  k_pool<<<cdiv(N_NODES, 256), 256, 0, stream>>>(h2, batch, pool, cnt, 128, N_NODES);
  k_gmean<<<cdiv(N_GRAPH * 128, 256), 256, 0, stream>>>(pool, cnt, g, 128, N_GRAPH);

  // --- MLP head (tiny: B=100) ---
  k_fc<<<cdiv(N_GRAPH * 256, 256), 256, 0, stream>>>(g, Wf1, bf1, t3, N_GRAPH, 128, 256);
  k_colstats<<<256, 256, 0, stream>>>(t3, N_GRAPH, 256, stats);
  k_bnrelu<<<cdiv(N_GRAPH * 256, 256), 256, 0, stream>>>(t3, t3, (_Float16*)nullptr,
                                                         stats, N_GRAPH, 256, 1);
  k_fc<<<cdiv(N_GRAPH * 128, 256), 256, 0, stream>>>(t3, Wf2, bf2, t4, N_GRAPH, 256, 128);
  k_colstats<<<128, 256, 0, stream>>>(t4, N_GRAPH, 128, stats);
  k_bnrelu<<<cdiv(N_GRAPH * 128, 256), 256, 0, stream>>>(t4, t4, (_Float16*)nullptr,
                                                         stats, N_GRAPH, 128, 1);
  k_fc<<<cdiv(N_GRAPH * 64, 256), 256, 0, stream>>>(t4, Wf3, bf3, (float*)d_out,
                                                    N_GRAPH, 128, 64);
  (void)in_sizes; (void)n_in; (void)out_size; (void)ws_size;
}
